// Bert_BiLSTM_CRF_29523605193150
// MI455X (gfx1250) — matrix operand order repaired and checked by reading, not compiled, not run on hardware
//
#include <hip/hip_runtime.h>
#include <hip/hip_bf16.h>

typedef __attribute__((ext_vector_type(16))) _Float16     v16h;
typedef __attribute__((ext_vector_type(8)))  float        v8f;
typedef __attribute__((ext_vector_type(4)))  unsigned int u32x4;
typedef __attribute__((ext_vector_type(4)))  float        f32x4;

#define NEG_INF_F (-10000.0f)
constexpr int B_ = 2048;
constexpr int T_ = 512;
constexpr int K_ = 24;
constexpr int WAVES = 8;   // waves per block; 8 blocks * 8 waves * 32 lanes = 2048 batches

// Forward CRF recurrence in exp domain via WMMA, plus per-batch gold score.
// One lane owns one batch chain (alpha[24] in VGPRs). Per step:
//   m = max(alpha); p = exp(alpha - m)  (f16, in [0,1])
//   S = P * E^T  via v_wmma_f32_16x16x32_f16 (K=32 covers 24 labels, zero-padded)
//   alpha = log(S) + m + feats[:,t,:]
__global__ __launch_bounds__(256) void crf_forward_kernel(
    const float* __restrict__ feats,      // [B,T,K]
    const float* __restrict__ trans,      // [K,K], trans[j,i] = score i->j
    const int*   __restrict__ tags,       // [B,T]
    const int*   __restrict__ start_ptr,  // [1]
    float*       __restrict__ ws)         // [B] per-batch (forward - gold)
{
  // wave-private LDS regions (strides keep 16B alignment + avoid bank conflicts)
  __shared__ __attribute__((aligned(16))) unsigned int p_lds[WAVES][32][20]; // f16 pairs, 80B stride
  __shared__ __attribute__((aligned(16))) float        s_lds[WAVES][32][36]; // f32 S,   144B stride
  __shared__ float trans_lds[K_ * K_];

  const int tid   = threadIdx.x;
  const int w     = tid >> 5;
  const int lane  = tid & 31;
  const int lhalf = lane & 15;
  const bool lo_half = (lane < 16);

  for (int i = tid; i < K_ * K_; i += 256) trans_lds[i] = trans[i];
  __syncthreads();

  const int start = start_ptr[0];
  const int b     = (blockIdx.x * WAVES + w) * 32 + lane;  // 0..2047 exactly

  // ---- Build constant B operand: E^T tiles in f16 WMMA B layout ----
  // B is 32x16 (KxN): lanes 0-15 hold K=0..15 (reg h -> K=h), lanes 16-31 hold K=16..31.
  // N = lane&15 -> label j = (lane&15) + 16*nt.  E[j][i] = exp(trans[j*24+i]), 0-padded.
  v16h Bop[2];
  #pragma unroll
  for (int nt = 0; nt < 2; ++nt) {
    const int j = lhalf + nt * 16;
    #pragma unroll
    for (int h = 0; h < 16; ++h) {
      const int i = (lo_half ? 0 : 16) + h;
      float e = 0.0f;
      if (j < K_ && i < K_) e = __expf(trans_lds[j * K_ + i]);
      Bop[nt][h] = (_Float16)e;
    }
  }

  float alpha[K_];
  #pragma unroll
  for (int j = 0; j < K_; ++j) alpha[j] = (j == start) ? 0.0f : NEG_INF_F;

  #pragma unroll 1
  for (int t = 1; t < T_; ++t) {
    // per-lane max over labels
    float m = alpha[0];
    #pragma unroll
    for (int j = 1; j < K_; ++j) m = fmaxf(m, alpha[j]);

    // pack p = exp(alpha - m) as 12 f16 pairs; store to wave-private LDS (batch-major)
    unsigned int dw[12];
    #pragma unroll
    for (int q = 0; q < 12; ++q) {
      union { _Float16 h[2]; unsigned int u; } pk;
      pk.h[0] = (_Float16)__expf(alpha[2 * q + 0] - m);
      pk.h[1] = (_Float16)__expf(alpha[2 * q + 1] - m);
      dw[q] = pk.u;
    }
    *(u32x4*)&p_lds[w][lane][0] = *(u32x4*)&dw[0];
    *(u32x4*)&p_lds[w][lane][4] = *(u32x4*)&dw[4];
    *(u32x4*)&p_lds[w][lane][8] = *(u32x4*)&dw[8];
    __syncthreads();

    // gather A tiles (f16 16x32 A layout) and run 4 WMMAs: 2 batch-tiles x 2 label-tiles
    // A layout: lanes 0-15: regs 0-3 -> K-pairs 0-3, regs 4-7 -> K-pairs 4-7
    //           lanes 16-31: regs 0-3 -> K-pairs 8-11, regs 4-7 -> K 24-31 (zero pad)
    v8f d[2][2];
    #pragma unroll
    for (int mt = 0; mt < 2; ++mt) {
      const int bl = mt * 16 + lhalf;
      const unsigned int* base = &p_lds[w][bl][0];
      union { v16h h; u32x4 q[2]; } au;
      au.q[0] = *(const u32x4*)(base + (lo_half ? 0 : 8));
      u32x4 hi4 = {};
      if (lo_half) hi4 = *(const u32x4*)(base + 4);
      au.q[1] = hi4;
      const v16h A = au.h;
      #pragma unroll
      for (int nt = 0; nt < 2; ++nt) {
        v8f c = {};
        d[mt][nt] = __builtin_amdgcn_wmma_f32_16x16x32_f16(
            false, A, false, Bop[nt], (short)0, c, false, false);
      }
    }

    // scatter D (f32 16x16 C/D layout) to S: lane<16 holds M=0..7, lane>=16 holds M=8..15
    #pragma unroll
    for (int mt = 0; mt < 2; ++mt) {
      #pragma unroll
      for (int nt = 0; nt < 2; ++nt) {
        const int jcol = lhalf + nt * 16;
        if (jcol < K_) {
          #pragma unroll
          for (int r = 0; r < 8; ++r) {
            const int bl = mt * 16 + (lo_half ? r : 8 + r);
            s_lds[w][bl][jcol] = d[mt][nt][r];
          }
        }
      }
    }
    __syncthreads();

    // gather per-batch S row + feats row; update alpha = log(S) + m + feat
    const size_t fidx = ((size_t)b * T_ + t) * K_;
    #pragma unroll
    for (int c4 = 0; c4 < 6; ++c4) {
      f32x4 sv = *(const f32x4*)&s_lds[w][lane][4 * c4];
      f32x4 fv = *(const f32x4*)&feats[fidx + 4 * c4];
      #pragma unroll
      for (int x = 0; x < 4; ++x)
        alpha[4 * c4 + x] = __logf(sv[x]) + m + fv[x];
    }
    if (t + 1 < T_) __builtin_prefetch(&feats[fidx + K_], 0, 0);
    __syncthreads();
  }

  // forward score = LSE over labels
  float mf = alpha[0];
  #pragma unroll
  for (int j = 1; j < K_; ++j) mf = fmaxf(mf, alpha[j]);
  float se = 0.0f;
  #pragma unroll
  for (int j = 0; j < K_; ++j) se += __expf(alpha[j] - mf);
  const float fwd = mf + __logf(se);

  // gold score: sum_t trans[tag_t, tag_{t-1}] + feats[b,t,tag_t]
  const size_t tbase = (size_t)b * T_;
  int prev = tags[tbase];
  float gold = 0.0f;
  #pragma unroll 1
  for (int t = 1; t < T_; ++t) {
    const int tg = tags[tbase + t];
    gold += trans_lds[tg * K_ + prev] + feats[(tbase + t) * (size_t)K_ + tg];
    prev = tg;
  }

  ws[b] = fwd - gold;
}

// Deterministic tree-reduce mean (no float atomics -> bitwise stable across replays)
__global__ __launch_bounds__(256) void crf_reduce_kernel(
    const float* __restrict__ ws, float* __restrict__ out)
{
  __shared__ float sm[256];
  float s = 0.0f;
  for (int i = threadIdx.x; i < B_; i += 256) s += ws[i];
  sm[threadIdx.x] = s;
  __syncthreads();
  for (int k = 128; k > 0; k >>= 1) {
    if ((int)threadIdx.x < k) sm[threadIdx.x] += sm[threadIdx.x + k];
    __syncthreads();
  }
  if (threadIdx.x == 0) out[0] = sm[0] * (1.0f / (float)B_);
}

extern "C" void kernel_launch(void* const* d_in, const int* in_sizes, int n_in,
                              void* d_out, int out_size, void* d_ws, size_t ws_size,
                              hipStream_t stream) {
  const float* feats = (const float*)d_in[0];
  const float* trans = (const float*)d_in[1];
  const int*   tags  = (const int*)d_in[2];
  const int*   start = (const int*)d_in[3];
  float* out = (float*)d_out;
  float* ws  = (float*)d_ws;

  crf_forward_kernel<<<dim3(B_ / (WAVES * 32)), dim3(256), 0, stream>>>(
      feats, trans, tags, start, ws);
  crf_reduce_kernel<<<dim3(1), dim3(256), 0, stream>>>(ws, out);
}
